// VGG16RoIHead_79817672229185
// MI455X (gfx1250) — compile-verified, matrix operand order checked by reading, not run on hardware
//
#include <hip/hip_runtime.h>
#include <hip/hip_bf16.h>

// ---------- types for WMMA ----------
typedef __attribute__((ext_vector_type(16))) __bf16 v16bf;
typedef __attribute__((ext_vector_type(8)))  float  v8f;

union BFrag { v16bf v; uint4 q[2]; };

// round-to-nearest-even f32 -> bf16 (inputs are finite; no NaN guard)
__device__ __forceinline__ unsigned short f32_to_bf16(float f) {
    unsigned int u = __float_as_uint(f);
    u += 0x7FFFu + ((u >> 16) & 1u);
    return (unsigned short)(u >> 16);
}

// pack bf16(lo) | bf16(hi) << 16
__device__ __forceinline__ unsigned int pack_bf16(float lo, float hi) {
    unsigned int ul = __float_as_uint(lo);
    unsigned int uh = __float_as_uint(hi);
    ul += 0x7FFFu + ((ul >> 16) & 1u);
    uh += 0x7FFFu + ((uh >> 16) & 1u);
    return (ul >> 16) | (uh & 0xFFFF0000u);
}

// =====================================================================
// RoI Align: per-RoI block. Selects FPN level, bilinear 2x2 sample avg,
// writes pooled features as bf16 row [c*49 + oy*7 + ox] of flat[512,25088].
// =====================================================================
__global__ __launch_bounds__(256)
void roi_pool_kernel(const float* __restrict__ f2,
                     const float* __restrict__ f3,
                     const float* __restrict__ f4,
                     const float* __restrict__ rois,
                     const int*   __restrict__ ridx,
                     unsigned short* __restrict__ flat)
{
    const int n = blockIdx.x;
    const float ry1 = rois[n*4+0], rx1 = rois[n*4+1];
    const float ry2 = rois[n*4+2], rx2 = rois[n*4+3];

    // level assignment: round(log2(sqrt(h*w)/224)+4), clip [2,4]
    const float rh_img = ry2 - ry1 + 1.0f;
    const float rw_img = rx2 - rx1 + 1.0f;
    float lvlf = rintf(log2f(sqrtf(rh_img * rw_img) / 224.0f) + 4.0f);
    lvlf = fminf(fmaxf(lvlf, 2.0f), 4.0f);
    const int lvl = (int)lvlf;

    const float* feat; int HW; float scale;
    if      (lvl == 2) { feat = f2; HW = 200; scale = 0.25f;   }
    else if (lvl == 3) { feat = f3; HW = 100; scale = 0.125f;  }
    else               { feat = f4; HW = 50;  scale = 0.0625f; }

    const int b = ridx[n];
    const float x1 = rx1 * scale, y1 = ry1 * scale;
    const float x2 = rx2 * scale, y2 = ry2 * scale;
    const float rw = fmaxf(x2 - x1, 1.0f);
    const float rh = fmaxf(y2 - y1, 1.0f);
    const float lim = (float)(HW - 1);

    for (int idx = threadIdx.x; idx < 512 * 49; idx += 256) {
        const int c   = idx / 49;
        const int pos = idx - c * 49;
        const int oy  = pos / 7;
        const int ox  = pos - oy * 7;
        const float* base = feat + ((size_t)b * 512 + c) * (size_t)(HW * HW);

        float sum = 0.0f;
        #pragma unroll
        for (int i = 0; i < 2; ++i) {
            float ys  = fminf(fmaxf(y1 + rh * ((float)(oy*2+i) + 0.5f) * (1.0f/14.0f), 0.0f), lim);
            float y0f = floorf(ys);
            float ly  = ys - y0f;
            int   yA  = (int)y0f;
            int   yB  = min(yA + 1, HW - 1);
            const float* r0 = base + (size_t)yA * HW;
            const float* r1 = base + (size_t)yB * HW;
            #pragma unroll
            for (int j = 0; j < 2; ++j) {
                float xs  = fminf(fmaxf(x1 + rw * ((float)(ox*2+j) + 0.5f) * (1.0f/14.0f), 0.0f), lim);
                float x0f = floorf(xs);
                float lx  = xs - x0f;
                int   xA  = (int)x0f;
                int   xB  = min(xA + 1, HW - 1);
                sum += r0[xA] * (1.0f-ly) * (1.0f-lx)
                     + r0[xB] * (1.0f-ly) * lx
                     + r1[xA] * ly        * (1.0f-lx)
                     + r1[xB] * ly        * lx;
            }
        }
        flat[(size_t)n * 25088 + idx] = f32_to_bf16(sum * 0.25f);
    }
}

// =====================================================================
// WMMA GEMM: C[M,Ntot] = (relu)(A_bf16[M,K] @ W_f32->bf16[K,Ntot] + bias)
// Block tile: 256 rows (grid.y) x 64 cols (grid.x). 8 waves; each wave
// owns two 16-row subtiles (offset 0 / 128) x four 16-col subtiles.
// K-step 64: W tile (64K x 64N f32) staged -> LDS bf16 transposed
// BT[n][k] with row stride 72 (16B-aligned fragments, conflict-reduced).
// Requires K % 64 == 0.
// =====================================================================
template<bool RELU, bool BF16OUT>
__global__ __launch_bounds__(256)
void gemm_wmma_kernel(const unsigned short* __restrict__ A,   // M x K bf16 row-major
                      const float* __restrict__ W,            // K x Ntot f32 row-major
                      const float* __restrict__ bias,         // Ntot
                      void* __restrict__ Cout,                 // M x Ntot (bf16 or f32)
                      int K, int Ntot)
{
    constexpr int LDB = 72;                       // BT row stride in bf16 elems
    __shared__ unsigned short BT[64 * LDB];       // [n][k] bf16, 9 KB

    const int tid  = threadIdx.x;
    const int lane = tid & 31;
    const int n0   = blockIdx.x * 64;
    const int m0   = blockIdx.y * 256 + (tid >> 5) * 16;  // subtile 0; subtile 1 at +128

    v8f acc[2][4] = {};

    // per-lane A addressing (16-bit A layout: lanes<16 -> K{0..7,16..23},
    // lanes>=16 -> K{8..15,24..31}, row = lane&15)
    const int kofl = (lane < 16) ? 0 : 8;
    const unsigned short* Arow0 = A + (size_t)(m0 +       (lane & 15)) * K + kofl;
    const unsigned short* Arow1 = A + (size_t)(m0 + 128 + (lane & 15)) * K + kofl;

    // staging map: thread -> (n group of 4 cols, k pair). Lanes sweep n ->
    // coalesced float4 global loads; packed dword LDS stores.
    const int ng = tid & 15;          // n = 4*ng .. 4*ng+3
    const int kp = tid >> 4;          // k pair base = 2*kp (covers 0..31)
    const bool full_tile = (n0 + 64 <= Ntot);

    for (int k0 = 0; k0 < K; k0 += 64) {
        // ---- stage W[k0..k0+63][n0..n0+63] -> BT (f32 -> bf16, transposed) ----
        if (full_tile) {
            #pragma unroll
            for (int r2 = 0; r2 < 64; r2 += 32) {
                const float* w0 = W + (size_t)(k0 + r2 + 2 * kp) * Ntot + n0 + 4 * ng;
                const float4 a = *(const float4*)w0;
                const float4 b = *(const float4*)(w0 + Ntot);
                __builtin_prefetch(w0 + (size_t)64 * Ntot, 0, 3);  // next k-tile
                unsigned int* dst = ((unsigned int*)BT)
                                  + (size_t)(4 * ng) * (LDB / 2) + (r2 >> 1) + kp;
                dst[0 * (LDB / 2)] = pack_bf16(a.x, b.x);
                dst[1 * (LDB / 2)] = pack_bf16(a.y, b.y);
                dst[2 * (LDB / 2)] = pack_bf16(a.z, b.z);
                dst[3 * (LDB / 2)] = pack_bf16(a.w, b.w);
            }
        } else {
            #pragma unroll
            for (int r2 = 0; r2 < 64; r2 += 32) {
                const float* w0 = W + (size_t)(k0 + r2 + 2 * kp) * Ntot + n0 + 4 * ng;
                #pragma unroll
                for (int i = 0; i < 4; ++i) {
                    const int  gn = n0 + 4 * ng + i;
                    const bool ok = gn < Ntot;
                    const float va = ok ? w0[i] : 0.0f;
                    const float vb = ok ? w0[Ntot + i] : 0.0f;
                    ((unsigned int*)BT)[(size_t)(4 * ng + i) * (LDB / 2) + (r2 >> 1) + kp]
                        = pack_bf16(va, vb);
                }
            }
        }
        __syncthreads();

        // ---- two 32-k WMMA chunks per staged tile ----
        #pragma unroll
        for (int kc = 0; kc < 64; kc += 32) {
            BFrag fa0, fa1;
            fa0.q[0] = *(const uint4*)(Arow0 + k0 + kc);
            fa0.q[1] = *(const uint4*)(Arow0 + k0 + kc + 16);
            fa1.q[0] = *(const uint4*)(Arow1 + k0 + kc);
            fa1.q[1] = *(const uint4*)(Arow1 + k0 + kc + 16);

            #pragma unroll
            for (int j = 0; j < 4; ++j) {
                BFrag fb;
                const unsigned short* bp = &BT[(j * 16 + (lane & 15)) * LDB + kc + kofl];
                fb.q[0] = *(const uint4*)(bp);
                fb.q[1] = *(const uint4*)(bp + 16);
                acc[0][j] = __builtin_amdgcn_wmma_f32_16x16x32_bf16(
                    false, fa0.v, false, fb.v, (short)0, acc[0][j], false, false);
                acc[1][j] = __builtin_amdgcn_wmma_f32_16x16x32_bf16(
                    false, fa1.v, false, fb.v, (short)0, acc[1][j], false, false);
            }
        }
        __syncthreads();
    }

    // ---- epilogue: C layout -> lane L holds (m = r + 8*(L>=16), n = L&15) ----
    const int mbase = m0 + ((lane >> 4) << 3);
    const int ncol0 = n0 + (lane & 15);
    #pragma unroll
    for (int s = 0; s < 2; ++s) {
        #pragma unroll
        for (int j = 0; j < 4; ++j) {
            const int gn = ncol0 + j * 16;
            if (gn >= Ntot) continue;
            const float bsum = bias[gn];
            #pragma unroll
            for (int r = 0; r < 8; ++r) {
                float o = acc[s][j][r] + bsum;
                if (RELU) o = fmaxf(o, 0.0f);
                const int gm = mbase + s * 128 + r;
                if (BF16OUT)
                    ((unsigned short*)Cout)[(size_t)gm * Ntot + gn] = f32_to_bf16(o);
                else
                    ((float*)Cout)[(size_t)gm * Ntot + gn] = o;
            }
        }
    }
}

// =====================================================================
// Launch
// =====================================================================
extern "C" void kernel_launch(void* const* d_in, const int* in_sizes, int n_in,
                              void* d_out, int out_size, void* d_ws, size_t ws_size,
                              hipStream_t stream) {
    const float* feat2   = (const float*)d_in[0];
    const float* feat3   = (const float*)d_in[1];
    const float* feat4   = (const float*)d_in[2];
    const float* rois    = (const float*)d_in[3];
    const int*   ridx    = (const int*)  d_in[4];
    const float* w_fc1   = (const float*)d_in[5];
    const float* b_fc1   = (const float*)d_in[6];
    const float* w_fc2   = (const float*)d_in[7];
    const float* b_fc2   = (const float*)d_in[8];
    const float* w_loc   = (const float*)d_in[9];
    const float* b_loc   = (const float*)d_in[10];
    const float* w_score = (const float*)d_in[11];
    const float* b_score = (const float*)d_in[12];

    const int N = 512, K1 = 25088, H = 4096, NL = 21 * 4, NS = 21;

    unsigned short* flat = (unsigned short*)d_ws;                 // 512 x 25088 bf16
    unsigned short* fc6  = flat + (size_t)N * K1;                 // 512 x 4096  bf16
    unsigned short* fc7  = fc6  + (size_t)N * H;                  // 512 x 4096  bf16
    float* out_loc   = (float*)d_out;                             // 512 x 84
    float* out_score = out_loc + (size_t)N * NL;                  // 512 x 21

    // 1) RoI Align -> bf16 features
    roi_pool_kernel<<<N, 256, 0, stream>>>(feat2, feat3, feat4, rois, ridx, flat);

    // 2) fc6 = relu(flat @ w_fc1 + b_fc1)   M=512,K=25088,N=4096
    gemm_wmma_kernel<true, true><<<dim3(H / 64, N / 256), 256, 0, stream>>>(
        flat, w_fc1, b_fc1, fc6, K1, H);

    // 3) fc7 = relu(fc6 @ w_fc2 + b_fc2)    M=512,K=4096,N=4096
    gemm_wmma_kernel<true, true><<<dim3(H / 64, N / 256), 256, 0, stream>>>(
        fc6, w_fc2, b_fc2, fc7, H, H);

    // 4) roi_cls_locs = fc7 @ w_loc + b_loc   N=84
    gemm_wmma_kernel<false, false><<<dim3((NL + 63) / 64, N / 256), 256, 0, stream>>>(
        fc7, w_loc, b_loc, out_loc, H, NL);

    // 5) roi_scores = fc7 @ w_score + b_score  N=21
    gemm_wmma_kernel<false, false><<<dim3((NS + 63) / 64, N / 256), 256, 0, stream>>>(
        fc7, w_score, b_score, out_score, H, NS);
}
